// MiniCPMAttention_39530878992859
// MI455X (gfx1250) — compile-verified, matrix operand order checked by reading, not run on hardware
//
#include <hip/hip_runtime.h>
#include <hip/hip_bf16.h>

// ---------------------------------------------------------------------------
// MiniCPM MLA attention, MI455X (gfx1250), wave32 + WMMA f32_16x16x32_f16,
// TDM tensor_load_to_lds for attention K-tile staging (when available).
// ---------------------------------------------------------------------------

typedef _Float16 v16h __attribute__((ext_vector_type(16)));
typedef _Float16 v8h  __attribute__((ext_vector_type(8)));
typedef float    v8f  __attribute__((ext_vector_type(8)));
typedef unsigned int v4u __attribute__((ext_vector_type(4)));
typedef int      v4i  __attribute__((ext_vector_type(4)));
typedef int      v8i  __attribute__((ext_vector_type(8)));

#define SEQ     2048
#define HIDDEN  2560
#define NHEADS  40
#define QLORA   768
#define KVLORA  256
#define ROPED   32
#define NOPED   64
#define VDIM    64
#define QHD     96     // NOPED + ROPED

#ifndef __has_builtin
#define __has_builtin(x) 0
#endif
#if __has_builtin(__builtin_amdgcn_tensor_load_to_lds) && \
    __has_builtin(__builtin_amdgcn_s_wait_tensorcnt)
#define USE_TDM 1
#else
#define USE_TDM 0
#endif

__device__ __forceinline__ v8f wmma_f16(v16h a, v16h b, v8f c) {
  return __builtin_amdgcn_wmma_f32_16x16x32_f16(false, a, false, b,
                                                (short)0, c, false, false);
}

// WMMA operand fragment from an LDS row (ISA 7.12.2, 16-bit 16x32):
// lane holds two contiguous 8-half chunks at k0=(lane/16)*8 and k0+16.
// rowp must be 16B-aligned (row strides are multiples of 16B).
__device__ __forceinline__ v16h ld_frag(const _Float16* rowp, int k0) {
  v8h lo = *(const v8h*)(rowp + k0);
  v8h hi = *(const v8h*)(rowp + k0 + 16);
  v16h r;
#pragma unroll
  for (int i = 0; i < 8; ++i) { r[i] = lo[i]; r[i + 8] = hi[i]; }
  return r;
}

#if USE_TDM
// Issue one TDM 2D tile load: tile_d1 rows x tile_d0 elements (2B each),
// row stride stride_d0 (elements), into packed LDS at lds_off.
// D# layout per CDNA5 ISA.
__device__ __forceinline__ void tdm_load_2d_f16(unsigned lds_off,
                                                const void* gptr,
                                                unsigned tile_d0,
                                                unsigned tile_d1,
                                                unsigned long long stride_d0) {
  unsigned long long ga = (unsigned long long)gptr;
  v4u g0;
  g0[0] = 1u;                                   // count=1, user descriptor
  g0[1] = lds_off;                              // lds_addr (bytes)
  g0[2] = (unsigned)(ga & 0xffffffffu);         // global_addr[31:0]
  g0[3] = (unsigned)((ga >> 32) & 0x1ffffffu)   // global_addr[56:32]
          | (2u << 30);                         // type=2 ("image")
  unsigned td0 = tile_d0;                       // tensor_dim0 == tile width
  unsigned td1 = 0x10000u;                      // tensor_dim1: large (no OOB)
  v8i g1;
  g1[0] = (int)(1u << 16);                      // data_size=1 (2 bytes)
  g1[1] = (int)((td0 & 0xffffu) << 16);         // tensor_dim0[15:0]
  g1[2] = (int)(((td0 >> 16) & 0xffffu) | ((td1 & 0xffffu) << 16));
  g1[3] = (int)(((td1 >> 16) & 0xffffu) | ((tile_d0 & 0xffffu) << 16));
  g1[4] = (int)(tile_d1 & 0xffffu);             // tile_dim1 (tile_dim2=0)
  g1[5] = (int)(unsigned)(stride_d0 & 0xffffffffu);   // dim0_stride lo
  g1[6] = (int)(unsigned)((stride_d0 >> 32) & 0xffffu); // dim0_stride hi
  g1[7] = 0;
  v4i gz = {0, 0, 0, 0};
#if __clang_major__ >= 23
  v8i gz8 = {0, 0, 0, 0, 0, 0, 0, 0};
  __builtin_amdgcn_tensor_load_to_lds(g0, g1, gz, gz, gz8, 0);
#else
  __builtin_amdgcn_tensor_load_to_lds(g0, g1, gz, gz, 0);
#endif
}
#endif  // USE_TDM

// ---------------------------------------------------------------------------
// Generic tiled WMMA GEMM: C[M,N] = A[M,K] * B[K,N]
// 256 threads (8 waves), tile 128x128, K-step 32, double-buffered LDS.
// Wave w: rows (w/2)*32, cols (w&1)*64 -> 2x4 = 8 wmma per K-step.
// Requires M % 128 == 0, K % 32 == 0; N guarded.
// ---------------------------------------------------------------------------
#define GBM 128
#define GBN 128
#define GBK 32
#define GPAD 16   // row stride (32+16)*2 = 96B, multiple of 16B

template <typename AT>
__device__ __forceinline__ void gemm_stage(const AT* __restrict__ A,
                                           const float* __restrict__ B,
                                           int N, int K, int bm, int bn, int k0,
                                           int tid,
                                           _Float16 (*As)[GBK + GPAD],
                                           _Float16 (*Bs)[GBK + GPAD]) {
#pragma unroll
  for (int i = 0; i < 4; ++i) {           // A tile 128x32 -> f16
    int r = (tid >> 3) + i * 32;
    int c = (tid & 7) * 4;
    const AT* src = A + (size_t)(bm + r) * K + k0 + c;
#pragma unroll
    for (int j = 0; j < 4; ++j) As[r][c + j] = (_Float16)src[j];
  }
#pragma unroll
  for (int i = 0; i < 4; ++i) {           // B tile 32x128 transposed, guarded
    int kr = (tid >> 5) + i * 8;
    int cn = (tid & 31) * 4;
    int gn = bn + cn;
    const float* src = B + (size_t)(k0 + kr) * N + gn;
#pragma unroll
    for (int j = 0; j < 4; ++j) {
      float v = (gn + j < N) ? src[j] : 0.0f;
      Bs[cn + j][kr] = (_Float16)v;
    }
  }
}

template <typename AT, typename CT>
__global__ __launch_bounds__(256)
void gemm_wmma(const AT* __restrict__ A, const float* __restrict__ B,
               CT* __restrict__ C, int M, int N, int K) {
  __shared__ _Float16 As[2][GBM][GBK + GPAD];
  __shared__ _Float16 Bs[2][GBN][GBK + GPAD];

  const int tid  = threadIdx.x;
  const int lane = tid & 31;
  const int wave = tid >> 5;
  const int bm = blockIdx.y * GBM;
  const int bn = blockIdx.x * GBN;
  const int wm = (wave >> 1) * 32;
  const int wn = (wave & 1) * 64;
  const int l16 = lane & 15;
  const int k0l = (lane >> 4) * 8;

  v8f acc[2][4];
#pragma unroll
  for (int mt = 0; mt < 2; ++mt)
#pragma unroll
    for (int nt = 0; nt < 4; ++nt)
#pragma unroll
      for (int r = 0; r < 8; ++r) acc[mt][nt][r] = 0.0f;

  const int nk = K / GBK;
  gemm_stage<AT>(A, B, N, K, bm, bn, 0, tid, As[0], Bs[0]);
  __syncthreads();

  for (int t = 0; t < nk; ++t) {
    const int cur = t & 1;
    if (t + 1 < nk) {
      gemm_stage<AT>(A, B, N, K, bm, bn, (t + 1) * GBK, tid, As[cur ^ 1],
                     Bs[cur ^ 1]);
      if (t + 2 < nk) {   // L2 prefetch two tiles ahead (global_prefetch_b8)
        __builtin_prefetch(A + (size_t)(bm + (tid >> 3)) * K + (t + 2) * GBK, 0, 0);
        __builtin_prefetch(B + (size_t)((t + 2) * GBK + (tid >> 5)) * N + bn +
                               (tid & 31) * 4, 0, 0);
      }
    }
    v16h af[2], bf[4];
#pragma unroll
    for (int mt = 0; mt < 2; ++mt)
      af[mt] = ld_frag(&As[cur][wm + mt * 16 + l16][0], k0l);
#pragma unroll
    for (int nt = 0; nt < 4; ++nt)
      bf[nt] = ld_frag(&Bs[cur][wn + nt * 16 + l16][0], k0l);
#pragma unroll
    for (int mt = 0; mt < 2; ++mt)
#pragma unroll
      for (int nt = 0; nt < 4; ++nt)
        acc[mt][nt] = wmma_f16(af[mt], bf[nt], acc[mt][nt]);
    __syncthreads();
  }

  // C layout: vgpr r -> row r + 8*(lane/16); col = lane%16 within 16-tile
#pragma unroll
  for (int mt = 0; mt < 2; ++mt)
#pragma unroll
    for (int nt = 0; nt < 4; ++nt)
#pragma unroll
      for (int r = 0; r < 8; ++r) {
        int row = bm + wm + mt * 16 + r + 8 * (lane >> 4);
        int col = bn + wn + nt * 16 + l16;
        if (col < N) C[(size_t)row * N + col] = (CT)acc[mt][nt][r];
      }
}

// ---------------------------------------------------------------------------
// RMSNorm over 768 (q_a path): f32 in -> f16 out
// ---------------------------------------------------------------------------
__global__ __launch_bounds__(256)
void rmsnorm_q_kernel(const float* __restrict__ x, const float* __restrict__ w,
                      _Float16* __restrict__ y) {
  const int s = blockIdx.x, t = threadIdx.x;
  const float* row = x + (size_t)s * QLORA;
  float v0 = row[t], v1 = row[t + 256], v2 = row[t + 512];
  float ss = v0 * v0 + v1 * v1 + v2 * v2;
#pragma unroll
  for (int off = 16; off >= 1; off >>= 1) ss += __shfl_xor(ss, off, 32);
  __shared__ float red[8];
  __shared__ float vsh;
  if ((t & 31) == 0) red[t >> 5] = ss;
  __syncthreads();
  if (t == 0) {
    float tot = 0.0f;
#pragma unroll
    for (int i = 0; i < 8; ++i) tot += red[i];
    vsh = tot / (float)QLORA;
  }
  __syncthreads();
  float inv = rsqrtf(vsh + 1e-6f);
  _Float16* yr = y + (size_t)s * QLORA;
  yr[t]       = (_Float16)(v0 * inv * w[t]);
  yr[t + 256] = (_Float16)(v1 * inv * w[t + 256]);
  yr[t + 512] = (_Float16)(v2 * inv * w[t + 512]);
}

__device__ __forceinline__ void rope_pair(float p, int i, float x, float other,
                                          float* out) {
  float freq = __powf(10000.0f, -(float)(i & 15) / 16.0f);
  float ang = p * freq;
  *out = x * __cosf(ang) + other * __sinf(ang);
}

// ---------------------------------------------------------------------------
// ckv post: rmsnorm first 256 -> f16; rope last 32 -> kpe f16
// ---------------------------------------------------------------------------
__global__ __launch_bounds__(256)
void ckv_post_kernel(const float* __restrict__ ckv, const float* __restrict__ w,
                     const int* __restrict__ pos,
                     _Float16* __restrict__ ckvn, _Float16* __restrict__ kpe) {
  const int s = blockIdx.x, t = threadIdx.x;
  const float* row = ckv + (size_t)s * (KVLORA + ROPED);
  float x = row[t];
  float ss = x * x;
#pragma unroll
  for (int off = 16; off >= 1; off >>= 1) ss += __shfl_xor(ss, off, 32);
  __shared__ float red[8];
  __shared__ float vsh;
  if ((t & 31) == 0) red[t >> 5] = ss;
  __syncthreads();
  if (t == 0) {
    float tot = 0.0f;
#pragma unroll
    for (int i = 0; i < 8; ++i) tot += red[i];
    vsh = tot / (float)KVLORA;
  }
  __syncthreads();
  float inv = rsqrtf(vsh + 1e-6f);
  ckvn[(size_t)s * KVLORA + t] = (_Float16)(x * inv * w[t]);
  if (t < ROPED) {
    float p = (float)pos[s];
    float xp = row[KVLORA + t];
    float other = (t < 16) ? -row[KVLORA + t + 16] : row[KVLORA + t - 16];
    float o;
    rope_pair(p, t, xp, other, &o);
    kpe[(size_t)s * ROPED + t] = (_Float16)o;
  }
}

// ---------------------------------------------------------------------------
// q rope + repack: q f32 [S][H][96] -> qf f16 [H][S][96]
// ---------------------------------------------------------------------------
__global__ void rope_q_kernel(const float* __restrict__ q,
                              const int* __restrict__ pos,
                              _Float16* __restrict__ qf) {
  const int s = blockIdx.x, h = blockIdx.y, t = threadIdx.x; // blockDim=96
  const float* src = q + ((size_t)s * NHEADS + h) * QHD;
  float v;
  if (t < NOPED) {
    v = src[t];
  } else {
    int i = t - NOPED;
    float p = (float)pos[s];
    float x = src[NOPED + i];
    float other = (i < 16) ? -src[NOPED + i + 16] : src[NOPED + i - 16];
    rope_pair(p, i, x, other, &v);
  }
  qf[((size_t)h * SEQ + s) * QHD + t] = (_Float16)v;
}

// ---------------------------------------------------------------------------
// Flash attention (causal). Grid: (S/128, NHEADS). Block: 256 (8 waves).
// Wave w owns q rows [qbase+16w, +16). K/V streamed in 64-key tiles.
// K tiles staged by TDM (wave 0 issues 2 descriptors, s_wait_tensorcnt,
// barrier publishes) while all waves stage V transposed.
// qf [H][S][96] f16 | kv [S][H][128] f16 (k_nope|v) | kpe [S][32] f16
// attn [S][H*64] f16
// ---------------------------------------------------------------------------
#define BQ   128
#define BKV  64
#define PROW 72    // 144B row stride (16B multiple)

__global__ __launch_bounds__(256)
void attn_kernel(const _Float16* __restrict__ qf,
                 const _Float16* __restrict__ kv,
                 const _Float16* __restrict__ kpe,
                 _Float16* __restrict__ attn) {
  __shared__ _Float16 Kn[BKV][NOPED];      // packed 128B rows (TDM dest)
  __shared__ _Float16 Kp[BKV][ROPED];      // packed 64B rows  (TDM dest)
  __shared__ _Float16 Vst[VDIM][PROW];     // V transposed: [vd][key]
  __shared__ _Float16 Ps[8][16][PROW];     // per-wave P tile (f16)

  const int tid  = threadIdx.x;
  const int lane = tid & 31;
  const int wave = tid >> 5;
  const int l16  = lane & 15;
  const int hi16 = lane >> 4;
  const int k0l  = hi16 * 8;
  const int h    = blockIdx.y;
  const int qbase = blockIdx.x * BQ;
  const int wrow  = qbase + wave * 16;

  // Q fragments straight from global (per-lane rows, 16B chunks)
  v16h aq[3];
  {
    const _Float16* qp = qf + ((size_t)h * SEQ + (wrow + l16)) * QHD;
#pragma unroll
    for (int kk = 0; kk < 3; ++kk) aq[kk] = ld_frag(qp, kk * 32 + k0l);
  }

  float m_r[8], l_r[8];
  v8f o[4];
#pragma unroll
  for (int r = 0; r < 8; ++r) { m_r[r] = -3.0e38f; l_r[r] = 0.0f; }
#pragma unroll
  for (int nt = 0; nt < 4; ++nt)
#pragma unroll
    for (int r = 0; r < 8; ++r) o[nt][r] = 0.0f;

  const float scale = rsqrtf((float)QHD);
  const int ktiles = (qbase + BQ) / BKV;   // causal bound for this block

  for (int kt = 0; kt < ktiles; ++kt) {
    const int kb = kt * BKV;
    __syncthreads();   // protect Kn/Kp/Vst from previous iteration's readers

#if USE_TDM
    // Wave 0 issues two TDM tile loads; TENSORcnt is per-wave, so wave 0
    // waits, then the barrier below publishes LDS to the block.
    if (wave == 0) {
      tdm_load_2d_f16((unsigned)(size_t)&Kn[0][0],
                      kv + ((size_t)kb * NHEADS + h) * 128,
                      NOPED, BKV, (unsigned long long)NHEADS * 128);
      tdm_load_2d_f16((unsigned)(size_t)&Kp[0][0],
                      kpe + (size_t)kb * ROPED,
                      ROPED, BKV, ROPED);
    }
#else
    for (int idx = tid; idx < BKV * 8; idx += 256) {   // k_nope 16B chunks
      int key = idx >> 3, c = idx & 7;
      *(uint4*)(&Kn[key][c * 8]) =
          *(const uint4*)(kv + ((size_t)(kb + key) * NHEADS + h) * 128 + c * 8);
    }
    for (int idx = tid; idx < BKV * 4; idx += 256) {   // k_pe 16B chunks
      int key = idx >> 2, c = idx & 3;
      *(uint4*)(&Kp[key][c * 8]) =
          *(const uint4*)(kpe + (size_t)(kb + key) * ROPED + c * 8);
    }
#endif
    // stage V transposed (overlaps the in-flight TDM loads)
    for (int idx = tid; idx < BKV * VDIM; idx += 256) {
      int key = idx >> 6, vd = idx & 63;
      Vst[vd][key] = kv[((size_t)(kb + key) * NHEADS + h) * 128 + 64 + vd];
    }
#if USE_TDM
    if (wave == 0) __builtin_amdgcn_s_wait_tensorcnt(0);
#endif
    __syncthreads();

    // S = Q K^T  (16x64 per wave): k_nope (kk=0,1) + k_pe (kk=2)
    v8f sa[4];
#pragma unroll
    for (int nt = 0; nt < 4; ++nt) {
#pragma unroll
      for (int r = 0; r < 8; ++r) sa[nt][r] = 0.0f;
      v16h b0 = ld_frag(&Kn[nt * 16 + l16][0], k0l);
      v16h b1 = ld_frag(&Kn[nt * 16 + l16][0], 32 + k0l);
      v16h b2 = ld_frag(&Kp[nt * 16 + l16][0], k0l);
      sa[nt] = wmma_f16(aq[0], b0, sa[nt]);
      sa[nt] = wmma_f16(aq[1], b1, sa[nt]);
      sa[nt] = wmma_f16(aq[2], b2, sa[nt]);
    }

    // online softmax (row = r + 8*hi16; 16 lanes hold 16 key columns)
    const int rowbase = wrow + 8 * hi16;
    const int keycol = kb + l16;
#pragma unroll
    for (int r = 0; r < 8; ++r) {
      int qr = rowbase + r;
      float mx = -3.0e38f;
#pragma unroll
      for (int nt = 0; nt < 4; ++nt) {
        float v = (keycol + nt * 16 <= qr) ? sa[nt][r] * scale : -3.0e38f;
        sa[nt][r] = v;
        mx = fmaxf(mx, v);
      }
#pragma unroll
      for (int off = 8; off >= 1; off >>= 1) mx = fmaxf(mx, __shfl_xor(mx, off, 32));
      float mnew = fmaxf(m_r[r], mx);
      float alpha = __expf(m_r[r] - mnew);
      float rsum = 0.0f;
#pragma unroll
      for (int nt = 0; nt < 4; ++nt) {
        float p = __expf(sa[nt][r] - mnew);
        sa[nt][r] = p;
        rsum += p;
      }
#pragma unroll
      for (int off = 8; off >= 1; off >>= 1) rsum += __shfl_xor(rsum, off, 32);
      l_r[r] = l_r[r] * alpha + rsum;
      m_r[r] = mnew;
#pragma unroll
      for (int nt = 0; nt < 4; ++nt) o[nt][r] *= alpha;
    }

    // write P tile to LDS as f16 (C layout -> row-major)
#pragma unroll
    for (int r = 0; r < 8; ++r) {
      int prow = r + 8 * hi16;
#pragma unroll
      for (int nt = 0; nt < 4; ++nt)
        Ps[wave][prow][nt * 16 + l16] = (_Float16)sa[nt][r];
    }
    __syncthreads();

    // O += P V
#pragma unroll
    for (int kk = 0; kk < 2; ++kk) {
      v16h ap = ld_frag(&Ps[wave][l16][0], kk * 32 + k0l);
#pragma unroll
      for (int nt = 0; nt < 4; ++nt) {
        v16h bv = ld_frag(&Vst[nt * 16 + l16][0], kk * 32 + k0l);
        o[nt] = wmma_f16(ap, bv, o[nt]);
      }
    }
  }

  // epilogue: normalize + store attn [S][H*64]
#pragma unroll
  for (int r = 0; r < 8; ++r) {
    int row = wrow + r + 8 * hi16;
    float inv = 1.0f / l_r[r];
#pragma unroll
    for (int nt = 0; nt < 4; ++nt)
      attn[(size_t)row * (NHEADS * VDIM) + h * VDIM + nt * 16 + l16] =
          (_Float16)(o[nt][r] * inv);
  }
}

// ---------------------------------------------------------------------------
// Host launch
// ---------------------------------------------------------------------------
extern "C" void kernel_launch(void* const* d_in, const int* in_sizes, int n_in,
                              void* d_out, int out_size, void* d_ws, size_t ws_size,
                              hipStream_t stream) {
  const float* hidden = (const float*)d_in[0];
  const int*   pos    = (const int*)d_in[1];
  const float* w_qa   = (const float*)d_in[2];
  const float* q_ln   = (const float*)d_in[3];
  const float* w_qb   = (const float*)d_in[4];
  const float* w_kva  = (const float*)d_in[5];
  const float* kv_ln  = (const float*)d_in[6];
  const float* w_kvb  = (const float*)d_in[7];
  const float* w_o    = (const float*)d_in[8];
  float* out = (float*)d_out;

  char* ws = (char*)d_ws;
  size_t off = 0;
  auto alloc = [&](size_t bytes) -> void* {
    void* p = ws + off;
    off = (off + bytes + 255) & ~(size_t)255;
    return p;
  };

  float*     qa     = (float*)alloc((size_t)SEQ * QLORA * 4);
  _Float16*  qan    = (_Float16*)alloc((size_t)SEQ * QLORA * 2);
  float*     q      = (float*)alloc((size_t)SEQ * NHEADS * QHD * 4);
  _Float16*  qfh    = (_Float16*)alloc((size_t)NHEADS * SEQ * QHD * 2);
  float*     ckv    = (float*)alloc((size_t)SEQ * (KVLORA + ROPED) * 4);
  _Float16*  ckvn   = (_Float16*)alloc((size_t)SEQ * KVLORA * 2);
  _Float16*  kpe    = (_Float16*)alloc((size_t)SEQ * ROPED * 2);
  _Float16*  kvbuf  = (_Float16*)alloc((size_t)SEQ * NHEADS * 128 * 2);
  _Float16*  attnb  = (_Float16*)alloc((size_t)SEQ * NHEADS * VDIM * 2);

  dim3 blk(256);

  // 1) qa = hidden @ w_qa
  gemm_wmma<float, float><<<dim3(QLORA / GBN, SEQ / GBM), blk, 0, stream>>>(
      hidden, w_qa, qa, SEQ, QLORA, HIDDEN);
  // 2) rmsnorm(qa) -> f16
  rmsnorm_q_kernel<<<dim3(SEQ), blk, 0, stream>>>(qa, q_ln, qan);
  // 3) q = qan @ w_qb
  gemm_wmma<_Float16, float><<<dim3((NHEADS * QHD) / GBN, SEQ / GBM), blk, 0, stream>>>(
      qan, w_qb, q, SEQ, NHEADS * QHD, QLORA);
  // 4) ckv = hidden @ w_kva  (N=288, guarded)
  gemm_wmma<float, float><<<dim3((KVLORA + ROPED + GBN - 1) / GBN, SEQ / GBM), blk, 0, stream>>>(
      hidden, w_kva, ckv, SEQ, KVLORA + ROPED, HIDDEN);
  // 5) rmsnorm(ckv[:256]) + rope(k_pe)
  ckv_post_kernel<<<dim3(SEQ), blk, 0, stream>>>(ckv, kv_ln, pos, ckvn, kpe);
  // 6) kv = ckvn @ w_kvb -> f16 [S][H][128]
  gemm_wmma<_Float16, _Float16><<<dim3((NHEADS * 128) / GBN, SEQ / GBM), blk, 0, stream>>>(
      ckvn, w_kvb, kvbuf, SEQ, NHEADS * 128, KVLORA);
  // 7) rope(q_pe) + repack head-major
  rope_q_kernel<<<dim3(SEQ, NHEADS), dim3(QHD), 0, stream>>>(q, pos, qfh);
  // 8) flash attention
  attn_kernel<<<dim3(SEQ / BQ, NHEADS), blk, 0, stream>>>(qfh, kvbuf, kpe, attnb);
  // 9) out = attn @ w_o
  gemm_wmma<_Float16, float><<<dim3(HIDDEN / GBN, SEQ / GBM), blk, 0, stream>>>(
      attnb, w_o, out, SEQ, HIDDEN, NHEADS * VDIM);
}
